// my_model_41394894798892
// MI455X (gfx1250) — matrix operand order repaired and checked by reading, not compiled, not run on hardware
//
#include <hip/hip_runtime.h>

// ---------------------------------------------------------------------------
// Types for CDNA5 WMMA (gfx1250, wave32)
// ---------------------------------------------------------------------------
typedef __bf16 bf16;
typedef __attribute__((ext_vector_type(16))) __bf16 v16bf;
typedef __attribute__((ext_vector_type(8)))  __bf16 v8bf;
typedef __attribute__((ext_vector_type(8)))  float  v8f;

#define B_  32      // batch
#define T_  256     // sequence length
#define E_  300     // embedding dim
#define EP_ 320     // padded embedding dim (10 * 32)
#define H_  512     // hidden
#define G4_ 2048    // 4*H
#define M_  256     // attention dim
#define S2_ 1024    // 2*H
#define BT_ 8192    // B*T

#define SCAN_THREADS 512   // 16 waves: 4 waves/SIMD -> ~256 VGPR budget, no spills
#define SCAN_WAVES   16

__device__ __forceinline__ float sigf(float x) { return 1.0f / (1.0f + __expf(-x)); }

// A-matrix fragment (16x32 bf16, row-major source, row = lane%16):
// lane-half h: elements 0..15 -> K = kb + 16h + e
__device__ __forceinline__ v16bf load_a_frag(const bf16* row, int kb, int hf) {
  union { v16bf v; v8bf h[2]; } u;
  u.h[0] = *(const v8bf*)(row + kb + hf * 16);
  u.h[1] = *(const v8bf*)(row + kb + hf * 16 + 8);
  return u.v;
}

// B-matrix fragment (32x16 bf16 = W^T tile): lane holds col N = lane%16,
// 16 consecutive K starting at kb + 16*(lane/16): ptr = W + n*ld + kb + 16*(lane/16)
__device__ __forceinline__ v16bf load_b_frag(const bf16* p) {
  union { v16bf v; v8bf h[2]; } u;
  u.h[0] = *(const v8bf*)(p);
  u.h[1] = *(const v8bf*)(p + 8);
  return u.v;
}

__device__ __forceinline__ v8f wmma_bf16(v16bf a, v16bf b, v8f c) {
  // (neg_a, A, neg_b, B, c_mod, C, reuse_a, reuse_b)
  return __builtin_amdgcn_wmma_f32_16x16x32_bf16(false, a, false, b, (short)0, c,
                                                 false, false);
}

// ---------------------------------------------------------------------------
// Weight convert / pad to bf16 (also slices column blocks out of fused weights)
// ---------------------------------------------------------------------------
__global__ void cvt_pad(const float* __restrict__ src, bf16* __restrict__ dst,
                        int rows, int src_ld, int col_off, int valid, int dst_ld) {
  const int total = rows * dst_ld;
  for (int i = blockIdx.x * blockDim.x + threadIdx.x; i < total;
       i += gridDim.x * blockDim.x) {
    const int r = i / dst_ld, c = i - r * dst_ld;
    float v = (c < valid) ? src[(size_t)r * src_ld + col_off + c] : 0.0f;
    dst[i] = (bf16)v;
  }
}

// ---------------------------------------------------------------------------
// Embedding gather -> bf16 (B*T, EP_) rows indexed by b*T+t
// ---------------------------------------------------------------------------
__global__ void embed_gather(const int* __restrict__ x, const float* __restrict__ embed,
                             bf16* __restrict__ embbf) {
  const int total = BT_ * EP_;
  for (int i = blockIdx.x * blockDim.x + threadIdx.x; i < total;
       i += gridDim.x * blockDim.x) {
    const int row = i / EP_, c = i - row * EP_;
    float v = (c < E_) ? embed[(size_t)x[row] * E_ + c] : 0.0f;
    embbf[i] = (bf16)v;
  }
}

// ---------------------------------------------------------------------------
// Persistent bidirectional LSTM: grid = 2 blocks (one per direction),
// 512 threads (16 waves), 4 tile-units per wave processed sequentially.
// State h (bf16, double-buffered) + c (f32) live in LDS.
// Per step: gates(32x2048) = x@Wih^T + h@Whh^T via WMMA, then cell update.
// ---------------------------------------------------------------------------
__global__ __launch_bounds__(SCAN_THREADS) void lstm_kernel(
    const bf16* __restrict__ emb,
    const bf16* __restrict__ WihF, const bf16* __restrict__ WhhF,
    const float* __restrict__ bF,
    const bf16* __restrict__ WihB, const bf16* __restrict__ WhhB,
    const float* __restrict__ bB,
    bf16* __restrict__ shared_out /* (B*T, 2H) bf16 */) {
  extern __shared__ char smem[];
  bf16*  hb0  = (bf16*)smem;             // 32*512 bf16
  bf16*  hb1  = hb0 + B_ * H_;           // 32*512 bf16
  float* c_st = (float*)(hb1 + B_ * H_); // 32*512 f32

  const int dir = blockIdx.x;
  const bf16*  Wih  = dir ? WihB : WihF;
  const bf16*  Whh  = dir ? WhhB : WhhF;
  const float* bias = dir ? bB : bF;

  const int tid  = threadIdx.x;
  const int wave = tid >> 5;
  const int lane = tid & 31;
  const int lrow = lane & 15;
  const int hf   = lane >> 4;

  for (int i = tid; i < B_ * H_; i += SCAN_THREADS) {
    hb0[i] = (bf16)0.0f; hb1[i] = (bf16)0.0f; c_st[i] = 0.0f;
  }
  __syncthreads();

  const v8f zero8 = {0.f, 0.f, 0.f, 0.f, 0.f, 0.f, 0.f, 0.f};

  for (int s = 0; s < T_; ++s) {
    const int t = dir ? (T_ - 1 - s) : s;
    bf16* hcur = (s & 1) ? hb1 : hb0;
    bf16* hnxt = (s & 1) ? hb0 : hb1;

    #pragma unroll 1   // keep one unit's pipeline live at a time (no spills)
    for (int u = 0; u < 4; ++u) {
      const int unit = wave * 4 + u;      // 64 units = 2 m-tiles x 32 col-tiles
      const int mt = unit >> 5;           // batch tile 0..1
      const int ct = unit & 31;           // hidden col tile 0..31
      const int abatch = mt * 16 + lrow;
      const bf16* ax = emb + (size_t)(abatch * T_ + t) * EP_;
      const bf16* ah = hcur + abatch * H_;

      v8f acc[4];
      #pragma unroll
      for (int g = 0; g < 4; ++g) acc[g] = zero8;

      for (int kb = 0; kb < EP_; kb += 32) {
        v16bf a = load_a_frag(ax, kb, hf);
        #pragma unroll
        for (int g = 0; g < 4; ++g) {
          const int n = g * H_ + ct * 16 + lrow;
          v16bf bm = load_b_frag(Wih + (size_t)n * EP_ + kb + hf * 16);
          acc[g] = wmma_bf16(a, bm, acc[g]);
        }
      }
      for (int kb = 0; kb < H_; kb += 32) {
        v16bf a = load_a_frag(ah, kb, hf);
        #pragma unroll
        for (int g = 0; g < 4; ++g) {
          const int n = g * H_ + ct * 16 + lrow;
          v16bf bm = load_b_frag(Whh + (size_t)n * H_ + kb + hf * 16);
          acc[g] = wmma_bf16(a, bm, acc[g]);
        }
      }

      const int j = ct * 16 + lrow;       // hidden index (C-layout N = lane%16)
      const float bi = bias[j], bfv = bias[H_ + j];
      const float bg = bias[2 * H_ + j], bo = bias[3 * H_ + j];
      #pragma unroll
      for (int r = 0; r < 8; ++r) {
        const int batch = mt * 16 + r + 8 * hf;   // C-layout M = r + 8*(lane/16)
        float iv = acc[0][r] + bi;
        float fv = acc[1][r] + bfv;
        float gv = acc[2][r] + bg;
        float ov = acc[3][r] + bo;
        float cold = c_st[batch * H_ + j];
        float cnew = sigf(fv) * cold + sigf(iv) * tanhf(gv);
        float hnew = sigf(ov) * tanhf(cnew);
        c_st[batch * H_ + j] = cnew;
        hnxt[batch * H_ + j] = (bf16)hnew;
        shared_out[(size_t)(batch * T_ + t) * S2_ + dir * H_ + j] = (bf16)hnew;
      }
    }
    __syncthreads();   // new h visible for next step (double-buffered => 1 sync)
  }
}

// ---------------------------------------------------------------------------
// base(B*T, M) = shared@Ws_s^T + emb@Ws_e^T + Ws_b  (time-invariant attention)
// One 16x16 tile per wave; 8 waves per block; 8192 tiles -> 1024 blocks.
// ---------------------------------------------------------------------------
__global__ __launch_bounds__(256) void base_kernel(
    const bf16* __restrict__ shbf, const bf16* __restrict__ emb,
    const bf16* __restrict__ WsS, const bf16* __restrict__ WsE,
    const float* __restrict__ Wsb, float* __restrict__ basep) {
  const int tid = threadIdx.x, wave = tid >> 5, lane = tid & 31;
  const int lrow = lane & 15, hf = lane >> 4;
  const int tile = blockIdx.x * 8 + wave;   // 0..8191
  const int mt = tile >> 4, nt = tile & 15;
  const int arow = mt * 16 + lrow;          // global row = b*T + t

  const v8f zero8 = {0.f, 0.f, 0.f, 0.f, 0.f, 0.f, 0.f, 0.f};
  v8f acc = zero8;

  const bf16* as = shbf + (size_t)arow * S2_;
  for (int kb = 0; kb < S2_; kb += 32) {
    v16bf a  = load_a_frag(as, kb, hf);
    v16bf bm = load_b_frag(WsS + (size_t)(nt * 16 + lrow) * S2_ + kb + hf * 16);
    acc = wmma_bf16(a, bm, acc);
  }
  const bf16* ae = emb + (size_t)arow * EP_;
  for (int kb = 0; kb < EP_; kb += 32) {
    v16bf a  = load_a_frag(ae, kb, hf);
    v16bf bm = load_b_frag(WsE + (size_t)(nt * 16 + lrow) * EP_ + kb + hf * 16);
    acc = wmma_bf16(a, bm, acc);
  }
  const float bb = Wsb[nt * 16 + lrow];
  #pragma unroll
  for (int r = 0; r < 8; ++r)
    basep[(size_t)(mt * 16 + r + 8 * hf) * M_ + nt * 16 + lrow] = acc[r] + bb;
}

// ---------------------------------------------------------------------------
// Persistent attentive decoder: 1 block, 512 threads (16 waves), state in LDS.
// Per step: hc = h@Ws_h^T (WMMA); Si = Us . tanh(base+hc); masked softmax;
// Rt = Bw . shared; gates = [emb|Rt]@cW^T + h@cU^T (WMMA); LSTM cell update.
// ---------------------------------------------------------------------------
__global__ __launch_bounds__(SCAN_THREADS) void decoder_kernel(
    const float* __restrict__ basep, const bf16* __restrict__ emb,
    const bf16* __restrict__ shbf,  const bf16* __restrict__ WsH,
    const float* __restrict__ Usw,  const float* __restrict__ Usb,
    const bf16* __restrict__ cWe,   const bf16* __restrict__ cWr,
    const bf16* __restrict__ cUb,   const float* __restrict__ cbv,
    const float* __restrict__ fcw,  const float* __restrict__ fcb,
    const int* __restrict__ mask,   float* __restrict__ out) {
  extern __shared__ char smem[];
  float* hc    = (float*)smem;            // 32*256 f32
  float* Bw    = hc + B_ * M_;            // 32*256 f32 (scores then weights)
  float* c_st  = Bw + B_ * T_;            // 32*512 f32
  float* usw_s = c_st + B_ * H_;          // 256 f32
  bf16*  hb0   = (bf16*)(usw_s + M_);     // 32*512 bf16
  bf16*  hb1   = hb0 + B_ * H_;           // 32*512 bf16
  bf16*  Rt    = hb1 + B_ * H_;           // 32*1024 bf16

  const int tid = threadIdx.x, wave = tid >> 5, lane = tid & 31;
  const int lrow = lane & 15, hf = lane >> 4;

  for (int i = tid; i < B_ * H_; i += SCAN_THREADS) {
    hb0[i] = (bf16)0.0f; hb1[i] = (bf16)0.0f; c_st[i] = 0.0f;
  }
  for (int i = tid; i < M_; i += SCAN_THREADS) usw_s[i] = Usw[i];
  __syncthreads();

  const v8f zero8 = {0.f, 0.f, 0.f, 0.f, 0.f, 0.f, 0.f, 0.f};

  for (int s = 0; s < T_; ++s) {
    const int t = s;
    bf16* hcur = (s & 1) ? hb1 : hb0;
    bf16* hnxt = (s & 1) ? hb0 : hb1;

    // --- 1. hc = h @ Ws_h^T : 32 tiles (2x16), 2 per wave -------------------
    #pragma unroll 1
    for (int tile = wave; tile < 32; tile += SCAN_WAVES) {
      const int mt = tile >> 4, nt = tile & 15;
      const int abatch = mt * 16 + lrow;
      v8f acc = zero8;
      for (int kb = 0; kb < H_; kb += 32) {
        v16bf a  = load_a_frag(hcur + abatch * H_, kb, hf);
        v16bf bm = load_b_frag(WsH + (size_t)(nt * 16 + lrow) * H_ + kb + hf * 16);
        acc = wmma_bf16(a, bm, acc);
      }
      #pragma unroll
      for (int r = 0; r < 8; ++r)
        hc[(mt * 16 + r + 8 * hf) * M_ + nt * 16 + lrow] = acc[r];
    }
    __syncthreads();

    // --- 2. Si[b,t'] = Us . tanh(base + hc) + Us_b (wave per pair) ----------
    #pragma unroll 1
    for (int p = wave; p < BT_; p += SCAN_WAVES) {
      const int b = p >> 8, tt = p & 255;
      const float* bp = basep + (size_t)p * M_;
      const float* hp = hc + b * M_;
      float partial = 0.0f;
      #pragma unroll
      for (int q = 0; q < 8; ++q) {
        const int m = lane + q * 32;
        partial += usw_s[m] * tanhf(bp[m] + hp[m]);
      }
      #pragma unroll
      for (int o = 16; o; o >>= 1) partial += __shfl_xor(partial, o, 32);
      if (lane == 0)
        Bw[p] = (mask[b * T_ + tt] > 0) ? (partial + Usb[0]) : -1e9f;
    }
    __syncthreads();

    // --- 3. softmax over T per batch row (2 rows per wave) ------------------
    #pragma unroll 1
    for (int b = wave; b < B_; b += SCAN_WAVES) {
      float vals[8]; float mx = -3.4e38f;
      #pragma unroll
      for (int q = 0; q < 8; ++q) {
        vals[q] = Bw[b * T_ + lane + q * 32];
        mx = fmaxf(mx, vals[q]);
      }
      #pragma unroll
      for (int o = 16; o; o >>= 1) mx = fmaxf(mx, __shfl_xor(mx, o, 32));
      float sum = 0.0f;
      #pragma unroll
      for (int q = 0; q < 8; ++q) { vals[q] = __expf(vals[q] - mx); sum += vals[q]; }
      #pragma unroll
      for (int o = 16; o; o >>= 1) sum += __shfl_xor(sum, o, 32);
      const float inv = 1.0f / sum;
      #pragma unroll
      for (int q = 0; q < 8; ++q) Bw[b * T_ + lane + q * 32] = vals[q] * inv;
    }
    __syncthreads();

    // --- 4. Rt[b,j] = sum_t Bw[b,t] * shared[b,t,j] -------------------------
    #pragma unroll 1
    for (int idx = tid; idx < B_ * S2_; idx += SCAN_THREADS) {
      const int b = idx >> 10, j = idx & (S2_ - 1);
      const bf16*  sp = shbf + (size_t)(b * T_) * S2_ + j;
      const float* wp = Bw + b * T_;
      float acc = 0.0f;
      for (int tt = 0; tt < T_; ++tt) acc += wp[tt] * (float)sp[(size_t)tt * S2_];
      Rt[idx] = (bf16)acc;
    }
    __syncthreads();

    // --- 5. gates = [emb|Rt]@cW^T + h@cU^T ; LSTM cell update ---------------
    #pragma unroll 1   // one unit pipeline live at a time
    for (int u = 0; u < 4; ++u) {
      const int unit = wave * 4 + u;
      const int mt = unit >> 5, ct = unit & 31;
      const int abatch = mt * 16 + lrow;
      const bf16* ax = emb + (size_t)(abatch * T_ + t) * EP_;
      const bf16* ar = Rt + abatch * S2_;
      const bf16* ah = hcur + abatch * H_;

      v8f acc[4];
      #pragma unroll
      for (int g = 0; g < 4; ++g) acc[g] = zero8;

      for (int kb = 0; kb < EP_; kb += 32) {
        v16bf a = load_a_frag(ax, kb, hf);
        #pragma unroll
        for (int g = 0; g < 4; ++g) {
          const int n = g * H_ + ct * 16 + lrow;
          v16bf bm = load_b_frag(cWe + (size_t)n * EP_ + kb + hf * 16);
          acc[g] = wmma_bf16(a, bm, acc[g]);
        }
      }
      for (int kb = 0; kb < S2_; kb += 32) {
        v16bf a = load_a_frag(ar, kb, hf);
        #pragma unroll
        for (int g = 0; g < 4; ++g) {
          const int n = g * H_ + ct * 16 + lrow;
          v16bf bm = load_b_frag(cWr + (size_t)n * S2_ + kb + hf * 16);
          acc[g] = wmma_bf16(a, bm, acc[g]);
        }
      }
      for (int kb = 0; kb < H_; kb += 32) {
        v16bf a = load_a_frag(ah, kb, hf);
        #pragma unroll
        for (int g = 0; g < 4; ++g) {
          const int n = g * H_ + ct * 16 + lrow;
          v16bf bm = load_b_frag(cUb + (size_t)n * H_ + kb + hf * 16);
          acc[g] = wmma_bf16(a, bm, acc[g]);
        }
      }

      const int j = ct * 16 + lrow;
      const float bi = cbv[j], bfv = cbv[H_ + j];
      const float bg = cbv[2 * H_ + j], bo = cbv[3 * H_ + j];
      #pragma unroll
      for (int r = 0; r < 8; ++r) {
        const int batch = mt * 16 + r + 8 * hf;
        float iv = acc[0][r] + bi;
        float fv = acc[1][r] + bfv;
        float gv = acc[2][r] + bg;
        float ov = acc[3][r] + bo;
        float cold = c_st[batch * H_ + j];
        float cnew = sigf(fv) * cold + sigf(iv) * tanhf(gv);
        float hnew = sigf(ov) * tanhf(cnew);
        c_st[batch * H_ + j] = cnew;
        hnxt[batch * H_ + j] = (bf16)hnew;
      }
    }
    __syncthreads();
  }

  // --- final: out = sigmoid(h_last @ fc_w^T + fc_b) -------------------------
  if (tid < B_) {
    const bf16* hp = hb0 + tid * H_;   // T_ even -> final state lives in hb0
    float a = fcb[0];
    for (int j = 0; j < H_; ++j) a += fcw[j] * (float)hp[j];
    out[tid] = sigf(a);
  }
}

// ---------------------------------------------------------------------------
// Host launcher
// ---------------------------------------------------------------------------
extern "C" void kernel_launch(void* const* d_in, const int* in_sizes, int n_in,
                              void* d_out, int out_size, void* d_ws, size_t ws_size,
                              hipStream_t stream) {
  (void)in_sizes; (void)n_in; (void)out_size; (void)ws_size;

  const int*   x     = (const int*)  d_in[0];
  const int*   mask  = (const int*)  d_in[1];
  // d_in[2] = TASK (unused by the reference computation)
  const float* embed = (const float*)d_in[3];
  const float* Wih_f = (const float*)d_in[4];
  const float* Whh_f = (const float*)d_in[5];
  const float* b_f   = (const float*)d_in[6];
  const float* Wih_b = (const float*)d_in[7];
  const float* Whh_b = (const float*)d_in[8];
  const float* b_b   = (const float*)d_in[9];
  const float* Ws_w  = (const float*)d_in[10];
  const float* Ws_b  = (const float*)d_in[11];
  const float* Us_w  = (const float*)d_in[12];
  const float* Us_b  = (const float*)d_in[13];
  const float* cW    = (const float*)d_in[14];
  const float* cU    = (const float*)d_in[15];
  const float* cb    = (const float*)d_in[16];
  const float* fc_w  = (const float*)d_in[17];
  const float* fc_b  = (const float*)d_in[18];
  float* out = (float*)d_out;

  char* ws = (char*)d_ws;
  size_t off = 0;
  auto take = [&](size_t bytes) -> void* {
    size_t o = (off + 255) & ~(size_t)255;
    off = o + bytes;
    return (void*)(ws + o);
  };

  bf16*  embbf = (bf16*) take((size_t)BT_ * EP_ * 2);
  bf16*  shbf  = (bf16*) take((size_t)BT_ * S2_ * 2);
  float* basep = (float*)take((size_t)BT_ * M_ * 4);
  bf16*  wihf  = (bf16*) take((size_t)G4_ * EP_ * 2);
  bf16*  whhf  = (bf16*) take((size_t)G4_ * H_ * 2);
  bf16*  wihb  = (bf16*) take((size_t)G4_ * EP_ * 2);
  bf16*  whhb  = (bf16*) take((size_t)G4_ * H_ * 2);
  bf16*  wss   = (bf16*) take((size_t)M_ * S2_ * 2);
  bf16*  wsh   = (bf16*) take((size_t)M_ * H_ * 2);
  bf16*  wse   = (bf16*) take((size_t)M_ * EP_ * 2);
  bf16*  cwe   = (bf16*) take((size_t)G4_ * EP_ * 2);
  bf16*  cwr   = (bf16*) take((size_t)G4_ * S2_ * 2);
  bf16*  cub   = (bf16*) take((size_t)G4_ * H_ * 2);

  // Weight conversion / padding / column-block slicing
  cvt_pad<<<1024, 256, 0, stream>>>(Wih_f, wihf, G4_, E_,        0,   E_,  EP_);
  cvt_pad<<<1024, 256, 0, stream>>>(Whh_f, whhf, G4_, H_,        0,   H_,  H_);
  cvt_pad<<<1024, 256, 0, stream>>>(Wih_b, wihb, G4_, E_,        0,   E_,  EP_);
  cvt_pad<<<1024, 256, 0, stream>>>(Whh_b, whhb, G4_, H_,        0,   H_,  H_);
  cvt_pad<<<1024, 256, 0, stream>>>(Ws_w,  wss,  M_,  3*H_+E_,   0,   S2_, S2_);
  cvt_pad<<<1024, 256, 0, stream>>>(Ws_w,  wsh,  M_,  3*H_+E_,   S2_, H_,  H_);
  cvt_pad<<<1024, 256, 0, stream>>>(Ws_w,  wse,  M_,  3*H_+E_,   3*H_, E_, EP_);
  cvt_pad<<<1024, 256, 0, stream>>>(cW,    cwe,  G4_, E_+S2_,    0,   E_,  EP_);
  cvt_pad<<<1024, 256, 0, stream>>>(cW,    cwr,  G4_, E_+S2_,    E_,  S2_, S2_);
  cvt_pad<<<1024, 256, 0, stream>>>(cU,    cub,  G4_, H_,        0,   H_,  H_);

  // Embedding gather (bf16, padded)
  embed_gather<<<2048, 256, 0, stream>>>(x, embed, embbf);

  // Bidirectional LSTM scan: 1 persistent workgroup per direction
  size_t lstm_lds = (size_t)(2 * B_ * H_ * 2 + B_ * H_ * 4);   // 128 KB
  lstm_kernel<<<2, SCAN_THREADS, lstm_lds, stream>>>(embbf, wihf, whhf, b_f,
                                                     wihb, whhb, b_b, shbf);

  // Hoisted attention base GEMM
  base_kernel<<<1024, 256, 0, stream>>>(shbf, embbf, wss, wse, Ws_b, basep);

  // Attentive decoder scan: 1 persistent workgroup
  size_t dec_lds = (size_t)(B_ * M_ * 4 + B_ * T_ * 4 + B_ * H_ * 4 + M_ * 4 +
                            2 * B_ * H_ * 2 + B_ * S2_ * 2);   // ~257 KB
  decoder_kernel<<<1, SCAN_THREADS, dec_lds, stream>>>(basep, embbf, shbf, wsh,
                                                       Us_w, Us_b, cwe, cwr, cub, cb,
                                                       fc_w, fc_b, mask, out);
}
